// PsychoFeatureExtractor_49271864820343
// MI455X (gfx1250) — matrix operand order repaired and checked by reading, not compiled, not run on hardware
//
#include <hip/hip_runtime.h>
#include <math.h>

// ---------------- problem constants ----------------
#define TSAMP   882000
#define HOP     513
#define NFFT    1024
#define NFRAMES 1720          // 1 + TSAMP/HOP
#define NBINS   360           // bins with freq <= 15500 Hz
#define NCOLS   720           // 2*NBINS (Re/Im interleaved)
#define NCOLP   768           // padded to multiple of 128
#define NBARK   24
#define FTILE   16            // frames per workgroup
#define XROW    (NFFT + 8)    // bf16 row stride (+8 elems = 16B pad -> conflict-free)
#define FSTEP   (22050.0f/512.0f) // bin -> Hz

typedef __bf16 v16bf __attribute__((ext_vector_type(16)));
typedef float  v8f   __attribute__((ext_vector_type(8)));

// ---------------- static device constants (filled by init kernels) ----------
__device__ __bf16 g_Dt[NCOLP * NFFT];   // windowed DFT, transposed: [col][k]
__device__ float  g_wk[NBINS];          // ear weight * (10^4.8 / winsum)^2
__device__ int    g_kstart[NBARK + 1];  // contiguous bin range per bark band
__device__ float  g_spread[NBARK * NBARK];

__constant__ float c_upper[NBARK] = {
    100.f, 200.f, 300.f, 400.f, 510.f, 630.f, 770.f, 920.f, 1080.f, 1270.f,
    1480.f, 1720.f, 2000.f, 2320.f, 2700.f, 3150.f, 3700.f, 4400.f, 5300.f,
    6400.f, 7700.f, 9500.f, 12000.f, 15500.f};

// ---------------- init: windowed DFT matrix (transposed) --------------------
__global__ __launch_bounds__(256) void psy_init_dt() {
    int idx = blockIdx.x * 256 + threadIdx.x;     // NCOLP*NFFT / 256 blocks
    int col = idx >> 10;
    int n   = idx & 1023;
    float v = 0.0f;
    if (col < NCOLS) {
        int k = col >> 1;
        // periodic hann
        float wn = 0.5f * (1.0f - __cosf(6.28318530717958647692f * (float)n * (1.0f / 1024.0f)));
        // exact angle reduction: (k*n) mod 1024
        int ph = (k * n) & 1023;
        float ang = 6.28318530717958647692f * (float)ph * (1.0f / 1024.0f);
        float s, c;
        __sincosf(ang, &s, &c);
        v = wn * ((col & 1) ? -s : c);            // rfft: Im = -sum x sin
    }
    g_Dt[idx] = (__bf16)v;
}

// ---------------- init: ear weights, band cuts, spreading -------------------
__global__ void psy_init_misc() {
    int t = threadIdx.x;
    if (t < NBINS) {
        float w;
        if (t == 0) {
            w = 1.0f;
        } else {
            float fk = (float)t * FSTEP * 0.001f;   // kHz
            float wadb = -3.64f * powf(fk, -0.8f)
                       + 6.5f * expf(-0.6f * (fk - 3.3f) * (fk - 3.3f))
                       - 0.001f * fk * fk * fk * fk;
            w = powf(10.0f, wadb * 0.1f);           // (10^(db/20))^2
        }
        // fold (10^(96/20))^2 / winsum^2, winsum = 512 exactly
        const float scl = (63095.734448019324f / 512.0f) * (63095.734448019324f / 512.0f);
        g_wk[t] = w * scl;
    }
    if (t < NBARK * NBARK) {
        int c = t / NBARK, d = t % NBARK;
        float bb = (float)(c - d) + 0.474f;
        float e  = 15.81f + 7.5f * bb - 17.5f * sqrtf(1.0f + bb * bb);
        g_spread[t] = powf(10.0f, e * 0.1f);
    }
    if (t == 0) {
        int k = 0;
        g_kstart[0] = 0;
        for (int c = 1; c <= NBARK; ++c) {
            while (k < NBINS && (float)k * FSTEP <= c_upper[c - 1]) k++;
            g_kstart[c] = k;
        }
    }
}

__device__ __forceinline__ long refl(long i) {
    return (i < 0) ? -i : ((i >= TSAMP) ? (2L * TSAMP - 2 - i) : i);
}

// ---------------- main: bf16 frame rows in LDS -> WMMA DFT -> bark tail -----
__global__ __launch_bounds__(256) void psy_main(const float* __restrict__ wav,
                                                float* __restrict__ out) {
    // 48 KB LDS, aliased: phase 1 = bf16 frame rows [16][XROW] (33 KB),
    //                     phase 2 = f32 Re/Im       [16][768]  (48 KB)
    __shared__ float smem[FTILE * NCOLP];

    const int f0  = blockIdx.x * FTILE;
    const int b   = blockIdx.y;
    const int tid = threadIdx.x;

    // ---- stage: de-overlap frames into bf16 rows (reflect padding) ----
    __bf16* xsf = (__bf16*)smem;
    const float* wrow = wav + (long)b * TSAMP;
    for (int p = tid; p < FTILE * 512; p += 256) {     // 2 samples per step
        const int f = p >> 9;
        const int n = (p & 511) << 1;
        const long i0 = (long)(f0 + f) * HOP + n - (NFFT / 2);
        xsf[f * XROW + n]     = (__bf16)wrow[refl(i0)];
        xsf[f * XROW + n + 1] = (__bf16)wrow[refl(i0 + 1)];
    }
    __syncthreads();

    const int wave = tid >> 5;
    const int lane = tid & 31;
    const int m    = lane & 15;      // frame row / column within group
    const int hi   = lane >> 4;
    const int nbase = wave * 96;     // 6 column sub-tiles of 16 per wave

    v8f acc[6];
#pragma unroll
    for (int t = 0; t < 6; ++t) acc[t] = (v8f)0.0f;

    for (int k0 = 0; k0 < NFFT; k0 += 32) {
        // A fragment: 16x32 bf16 from LDS frame rows, 2 x b128 per lane.
        // lanes 0-15: K = k0+0..15 ; lanes 16-31: K = k0+16..31
        const int kA = k0 + (hi ? 16 : 0);
        const __bf16* ap = xsf + m * XROW + kA;        // 16B aligned
        v16bf a;
        ((uint4*)&a)[0] = *(const uint4*)ap;
        ((uint4*)&a)[1] = *(const uint4*)(ap + 8);

        // B fragments: 32x16 bf16 from g_Dt (contiguous in K per column).
        // lanes 0-15: K = k0+0..15 ; lanes 16-31: K = k0+16..31
        const int kB = k0 + (hi ? 16 : 0);
#pragma unroll
        for (int t = 0; t < 6; ++t) {
            const int col = nbase + t * 16 + m;
            const __bf16* bp = g_Dt + col * NFFT + kB; // 32B aligned
            v16bf bf;
            ((uint4*)&bf)[0] = ((const uint4*)bp)[0];
            ((uint4*)&bf)[1] = ((const uint4*)bp)[1];
            acc[t] = __builtin_amdgcn_wmma_f32_16x16x32_bf16(
                false, a, false, bf, (short)0, acc[t], false, false);
        }
    }
    __syncthreads();   // everyone done reading xsf before aliasing it

    // ---- C tiles -> LDS ReIm[frame][col]; VGPR r: M=r(+8 for hi lanes), N=lane&15
#pragma unroll
    for (int t = 0; t < 6; ++t) {
        const int n = nbase + t * 16 + m;
        const int rowb = hi ? 8 : 0;
#pragma unroll
        for (int r = 0; r < 8; ++r)
            smem[(rowb + r) * NCOLP + n] = acc[t][r];
    }
    __syncthreads();

    // ---- scalar tail: bark bands -> spreading -> dB -> sone -> ntot ----
    if (tid < FTILE) {
        const int f = f0 + tid;
        if (f < NFRAMES) {
            const float* row = smem + tid * NCOLP;
            float bark[NBARK];
#pragma unroll
            for (int c = 0; c < NBARK; ++c) {
                float s = 0.0f;
                const int ke = g_kstart[c + 1];
                for (int k = g_kstart[c]; k < ke; ++k) {
                    float re = row[2 * k], im = row[2 * k + 1];
                    s += (re * re + im * im) * g_wk[k];
                }
                bark[c] = s;
            }
            float mx = 0.0f, sm = 0.0f;
#pragma unroll
            for (int c = 0; c < NBARK; ++c) {
                float sp = 0.0f;
#pragma unroll
                for (int d = 0; d < NBARK; ++d)
                    sp += g_spread[c * NBARK + d] * bark[d];
                float db = 10.0f * log10f(fmaxf(sp, 1.0f));
                float sone = (db >= 40.0f) ? exp2f((db - 40.0f) * 0.1f)
                                           : powf(db * 0.025f, 2.642f);
                sm += sone;
                mx = fmaxf(mx, sone);
            }
            out[b * NFRAMES + f] = mx + 0.15f * (sm - mx);
        }
    }
}

// ---------------- launch ----------------------------------------------------
extern "C" void kernel_launch(void* const* d_in, const int* in_sizes, int n_in,
                              void* d_out, int out_size, void* d_ws, size_t ws_size,
                              hipStream_t stream) {
    (void)in_sizes; (void)n_in; (void)out_size; (void)d_ws; (void)ws_size;
    const float* wav = (const float*)d_in[0];
    float* out = (float*)d_out;

    psy_init_dt<<<dim3((NCOLP * NFFT) / 256), dim3(256), 0, stream>>>();
    psy_init_misc<<<dim3(1), dim3(NBARK * NBARK), 0, stream>>>();

    const int ftiles = (NFRAMES + FTILE - 1) / FTILE;   // 108
    psy_main<<<dim3(ftiles, 32), dim3(256), 0, stream>>>(wav, out);
}
